// NasCiteseerCell_41480794145002
// MI455X (gfx1250) — matrix operand order repaired and checked by reading, not compiled, not run on hardware
//
#include <hip/hip_runtime.h>
#include <stdint.h>

#define NNODES 50000
#define NEDGES 800000
#define CURD   512
#define HIDD   64
#define OUTD   64
#define HEADS  6
#define HO     (HEADS*OUTD)   /* 384 */
#define CATD   (HIDD+HO)      /* 448 */

typedef __attribute__((ext_vector_type(16))) __bf16 v16bf;
typedef __attribute__((ext_vector_type(8)))  float  v8f;

static __device__ __forceinline__ float bf2f(unsigned short u){
  return __uint_as_float(((unsigned)u) << 16);
}
static __device__ __forceinline__ unsigned short f2bf(float f){
  unsigned x = __float_as_uint(f);
  unsigned r = x + 0x7FFFu + ((x >> 16) & 1u);
  return (unsigned short)(r >> 16);
}
static __device__ __forceinline__ float lrelu(float x, float s){ return x > 0.f ? x : s * x; }

// Build a v16bf fragment from two 16-byte runs (vector loads, no per-half moves).
static __device__ __forceinline__ v16bf load_a2(const unsigned short* p0, const unsigned short* p1){
  union { uint4 q[2]; v16bf v; } u;
  u.q[0] = *(const uint4*)p0;
  u.q[1] = *(const uint4*)p1;
  return u.v;
}
static __device__ __forceinline__ v16bf load_frag16(const unsigned short* p){
  return load_a2(p, p + 8);
}
// f32 row -> bf16 A fragment (two 32-byte runs), converts via v_cvt_pk_bf16_f32.
static __device__ __forceinline__ v16bf load_a_f32(const float* p0, const float* p1){
  float4 f0 = *(const float4*)p0;
  float4 f1 = *(const float4*)(p0 + 4);
  float4 f2 = *(const float4*)p1;
  float4 f3 = *(const float4*)(p1 + 4);
  v16bf a;
  a[0]=(__bf16)f0.x;  a[1]=(__bf16)f0.y;  a[2]=(__bf16)f0.z;  a[3]=(__bf16)f0.w;
  a[4]=(__bf16)f1.x;  a[5]=(__bf16)f1.y;  a[6]=(__bf16)f1.z;  a[7]=(__bf16)f1.w;
  a[8]=(__bf16)f2.x;  a[9]=(__bf16)f2.y;  a[10]=(__bf16)f2.z; a[11]=(__bf16)f2.w;
  a[12]=(__bf16)f3.x; a[13]=(__bf16)f3.y; a[14]=(__bf16)f3.z; a[15]=(__bf16)f3.w;
  return a;
}
static __device__ __forceinline__ v8f wmma_bf16(v16bf a, v16bf b, v8f c){
  return __builtin_amdgcn_wmma_f32_16x16x32_bf16(false, a, false, b, (short)0, c, false, false);
}

// Stage `bytes` of global data into LDS with the CDNA5 async-to-LDS path.
// Every thread of the block must reach this (contains a barrier).
static __device__ __forceinline__ void stage_lds(unsigned short* lds, const unsigned short* g, int bytes){
  int tid = threadIdx.x;
  unsigned lbase = (unsigned)(uintptr_t)lds;          // low 32 bits of generic ptr = LDS byte addr
  for (int off = tid * 16; off < bytes; off += (int)blockDim.x * 16){
    unsigned long long ga = (unsigned long long)(uintptr_t)((const char*)g + off);
    unsigned la = lbase + off;
    asm volatile("global_load_async_to_lds_b128 %0, %1, off" :: "v"(la), "v"(ga) : "memory");
  }
  asm volatile("s_wait_asynccnt 0x0" ::: "memory");
  __syncthreads();
}

// ---------------- K0: pre-swizzle weights into B-fragment layout (bf16) ----------------
// tile t: index (kb * nnb + nb); layout [tile][lane(32)][e(16)],
// value = W[(kb*32 + 8*(lane>>4) + (e<8 ? e : e+8)) * NC + nb*16 + (lane&15)]
__global__ void k_swz_weights(const float* __restrict__ Wpre, const float* __restrict__ Wgat,
                              const float* __restrict__ Wlin,
                              unsigned short* __restrict__ spre, unsigned short* __restrict__ sgat,
                              unsigned short* __restrict__ slin)
{
  int tile = blockIdx.x;           // 0..167
  int lane = threadIdx.x;          // 0..31
  const float* W; unsigned short* dst; int NC; int t;
  if (tile < 64)       { W = Wpre; dst = spre; NC = 64;  t = tile;       }  // 16 kb x 4 nb
  else if (tile < 112) { W = Wgat; dst = sgat; NC = 384; t = tile - 64;  }  // 2 kb x 24 nb
  else                 { W = Wlin; dst = slin; NC = 64;  t = tile - 112; }  // 14 kb x 4 nb
  int nnb = NC / 16;
  int kb = t / nnb, nb = t % nnb;
  int koff = 8 * (lane >> 4);
  int n = nb * 16 + (lane & 15);
  unsigned short* o = dst + ((size_t)t * 32 + lane) * 16;
  #pragma unroll
  for (int e = 0; e < 16; ++e){
    int k = kb * 32 + koff + (e < 8 ? e : e + 8);
    o[e] = f2bf(W[(size_t)k * NC + n]);
  }
}

// ---------------- K1: h = bf16(x @ W_pre + b_pre), one wave = 16 rows x 64 cols ----------------
__global__ void k_pre_gemm(const float* __restrict__ x, const unsigned short* __restrict__ swz,
                           const float* __restrict__ bpre, unsigned short* __restrict__ h)
{
  __shared__ __align__(16) unsigned short sw[16384];   // 32 KB: first half of W_pre swizzled (kb 0..7)
  stage_lds(sw, swz, 32768);

  int gid = blockIdx.x * blockDim.x + threadIdx.x;
  int w = gid >> 5, lane = gid & 31;
  if (w >= NNODES / 16) return;
  int row0 = w * 16;
  int nl = lane & 15, hi = lane >> 4;
  int koff = 8 * hi;
  const float* rowp = x + (size_t)(row0 + nl) * CURD;
  v8f a0 = {}, a1 = {}, a2 = {}, a3 = {};
  #pragma unroll 4
  for (int kb = 0; kb < CURD / 32; ++kb){
    v16bf A = load_a_f32(rowp + kb * 32 + koff, rowp + kb * 32 + 16 + koff);
    const unsigned short* bb = sw + (kb * 4) * 512 + lane * 16;
    v16bf B0 = load_frag16(bb);
    v16bf B1 = load_frag16(bb + 512);
    v16bf B2 = load_frag16(bb + 1024);
    v16bf B3 = load_frag16(bb + 1536);
    a0 = wmma_bf16(A, B0, a0);
    a1 = wmma_bf16(A, B1, a1);
    a2 = wmma_bf16(A, B2, a2);
    a3 = wmma_bf16(A, B3, a3);
  }
  #pragma unroll
  for (int r = 0; r < 8; ++r){
    int row = row0 + r + 8 * hi;
    unsigned short* op = h + (size_t)row * HIDD;
    op[nl]      = f2bf(a0[r] + bpre[nl]);
    op[16 + nl] = f2bf(a1[r] + bpre[16 + nl]);
    op[32 + nl] = f2bf(a2[r] + bpre[32 + nl]);
    op[48 + nl] = f2bf(a3[r] + bpre[48 + nl]);
  }
}

// ---------------- K2: xh = bf16(h @ W_gat), + per-row attention dots a_src/a_dst ----------------
// one wave = 16 rows x one head (64 cols)
__global__ void k_gat_gemm(const unsigned short* __restrict__ h, const unsigned short* __restrict__ swz,
                           const float* __restrict__ attS, const float* __restrict__ attD,
                           unsigned short* __restrict__ xh, float* __restrict__ asrc, float* __restrict__ adst)
{
  __shared__ __align__(16) unsigned short sw[24576];   // 48 KB: all of W_gat swizzled
  stage_lds(sw, swz, 49152);

  int gid = blockIdx.x * blockDim.x + threadIdx.x;
  int w = gid >> 5, lane = gid & 31;
  const int NW = (NNODES / 16) * HEADS;
  if (w >= NW) return;
  int hd = w % HEADS, rt = w / HEADS;
  int row0 = rt * 16;
  int nl = lane & 15, hi = lane >> 4;
  int koff = 8 * hi;
  const unsigned short* rowp = h + (size_t)(row0 + nl) * HIDD;
  v8f a0 = {}, a1 = {}, a2 = {}, a3 = {};
  #pragma unroll
  for (int kb = 0; kb < 2; ++kb){
    v16bf A = load_a2(rowp + kb * 32 + koff, rowp + kb * 32 + 16 + koff);
    const unsigned short* bb = sw + (kb * 24 + hd * 4) * 512 + lane * 16;
    v16bf B0 = load_frag16(bb);
    v16bf B1 = load_frag16(bb + 512);
    v16bf B2 = load_frag16(bb + 1024);
    v16bf B3 = load_frag16(bb + 1536);
    a0 = wmma_bf16(A, B0, a0);
    a1 = wmma_bf16(A, B1, a1);
    a2 = wmma_bf16(A, B2, a2);
    a3 = wmma_bf16(A, B3, a3);
  }
  float ws0 = attS[hd * 64 + nl],      wd0 = attD[hd * 64 + nl];
  float ws1 = attS[hd * 64 + 16 + nl], wd1 = attD[hd * 64 + 16 + nl];
  float ws2 = attS[hd * 64 + 32 + nl], wd2 = attD[hd * 64 + 32 + nl];
  float ws3 = attS[hd * 64 + 48 + nl], wd3 = attD[hd * 64 + 48 + nl];
  #pragma unroll
  for (int r = 0; r < 8; ++r){
    int row = row0 + r + 8 * hi;
    unsigned short* op = xh + (size_t)row * HO + hd * OUTD;
    op[nl]      = f2bf(a0[r]);
    op[16 + nl] = f2bf(a1[r]);
    op[32 + nl] = f2bf(a2[r]);
    op[48 + nl] = f2bf(a3[r]);
    float s = a0[r] * ws0 + a1[r] * ws1 + a2[r] * ws2 + a3[r] * ws3;
    float d = a0[r] * wd0 + a1[r] * wd1 + a2[r] * wd2 + a3[r] * wd3;
    #pragma unroll
    for (int m = 8; m >= 1; m >>= 1){
      s += __shfl_xor(s, m, 32);
      d += __shfl_xor(d, m, 32);
    }
    if (nl == 0){
      asrc[(size_t)row * HEADS + hd] = s;
      adst[(size_t)row * HEADS + hd] = d;
    }
  }
}

// ---------------- edge softmax passes ----------------
__device__ __forceinline__ void atomicMaxF(float* p, float v){
  int cur = __float_as_int(*p);
  while (__int_as_float(cur) < v){
    int assumed = cur;
    cur = atomicCAS((int*)p, assumed, __float_as_int(v));
    if (cur == assumed) break;
  }
}

__global__ void k_self_max(const float* __restrict__ asrc, const float* __restrict__ adst,
                           float* __restrict__ maxv)
{
  int i = blockIdx.x * blockDim.x + threadIdx.x;
  if (i >= NNODES * HEADS) return;
  maxv[i] = lrelu(asrc[i] + adst[i], 0.2f);   // self-loop alpha initializes the max
}

__global__ void k_edge_max(const long long* __restrict__ ei, const float* __restrict__ asrc,
                           const float* __restrict__ adst, float* __restrict__ maxv)
{
  int t = blockIdx.x * blockDim.x + threadIdx.x;
  if (t >= NEDGES * HEADS) return;
  int e = t / HEADS, hd = t % HEADS;
  int s = (int)ei[e], d = (int)ei[NEDGES + e];
  float a = lrelu(asrc[s * HEADS + hd] + adst[d * HEADS + hd], 0.2f);
  atomicMaxF(&maxv[d * HEADS + hd], a);
}

__global__ void k_self_den(const float* __restrict__ asrc, const float* __restrict__ adst,
                           const float* __restrict__ maxv, float* __restrict__ den)
{
  int i = blockIdx.x * blockDim.x + threadIdx.x;
  if (i >= NNODES * HEADS) return;
  float a = lrelu(asrc[i] + adst[i], 0.2f);
  den[i] = __expf(a - maxv[i]);
}

__global__ void k_edge_den(const long long* __restrict__ ei, const float* __restrict__ asrc,
                           const float* __restrict__ adst, const float* __restrict__ maxv,
                           float* __restrict__ den)
{
  int t = blockIdx.x * blockDim.x + threadIdx.x;
  if (t >= NEDGES * HEADS) return;
  int e = t / HEADS, hd = t % HEADS;
  int s = (int)ei[e], d = (int)ei[NEDGES + e];
  int i = d * HEADS + hd;
  float a = lrelu(asrc[s * HEADS + hd] + adst[i], 0.2f);
  atomicAdd(&den[i], __expf(a - maxv[i]));
}

// self-loop contribution initializes the bf16 accumulator (one write per pair)
__global__ void k_self_scatter(const float* __restrict__ asrc, const float* __restrict__ adst,
                               const float* __restrict__ maxv, const float* __restrict__ den,
                               const unsigned short* __restrict__ xh, unsigned short* __restrict__ accb)
{
  int t = blockIdx.x * blockDim.x + threadIdx.x;
  if (t >= NNODES * (HO / 2)) return;
  int n = t / (HO / 2), j = t % (HO / 2);
  int c = j * 2, hd = c / OUTD;
  int i = n * HEADS + hd;
  float a = lrelu(asrc[i] + adst[i], 0.2f);
  float coef = __expf(a - maxv[i]) / den[i];
  unsigned u = *(const unsigned*)(xh + (size_t)n * HO + c);
  float lo = bf2f((unsigned short)u) * coef;
  float hv = bf2f((unsigned short)(u >> 16)) * coef;
  *(unsigned*)(accb + (size_t)n * HO + c) = (unsigned)f2bf(lo) | ((unsigned)f2bf(hv) << 16);
}

// one wave per (edge, head); lane handles 2 channels; packed bf16 atomic add
__global__ void k_edge_scatter(const long long* __restrict__ ei, const float* __restrict__ asrc,
                               const float* __restrict__ adst, const float* __restrict__ maxv,
                               const float* __restrict__ den, const unsigned short* __restrict__ xh,
                               unsigned short* __restrict__ accb)
{
  int gid = blockIdx.x * blockDim.x + threadIdx.x;
  int w = gid >> 5, lane = gid & 31;
  if (w >= NEDGES * HEADS) return;
  int e = w / HEADS, hd = w % HEADS;
  int s = (int)ei[e], d = (int)ei[NEDGES + e];
  int i = d * HEADS + hd;
  float a = lrelu(asrc[s * HEADS + hd] + adst[i], 0.2f);
  float coef = __expf(a - maxv[i]) / den[i];
  int c = hd * OUTD + lane * 2;
  unsigned u = *(const unsigned*)(xh + (size_t)s * HO + c);
  float lo = bf2f((unsigned short)u) * coef;
  float hv = bf2f((unsigned short)(u >> 16)) * coef;
  unsigned vdat = (unsigned)f2bf(lo) | ((unsigned)f2bf(hv) << 16);
  unsigned long long addr = (unsigned long long)(accb + (size_t)d * HO + c);
  asm volatile("global_atomic_pk_add_bf16 %0, %1, off" :: "v"(addr), "v"(vdat) : "memory");
}

// ---------------- K9: out = concat(h, leaky(acc+b_gat,0.01)) @ W_lin + b_lin ----------------
__global__ void k_final_gemm(const unsigned short* __restrict__ h, const unsigned short* __restrict__ accb,
                             const float* __restrict__ bgat, const unsigned short* __restrict__ swz,
                             const float* __restrict__ blin, float* __restrict__ out)
{
  __shared__ __align__(16) unsigned short sw[28672];   // 56 KB: all of W_lin swizzled
  stage_lds(sw, swz, 57344);

  int gid = blockIdx.x * blockDim.x + threadIdx.x;
  int w = gid >> 5, lane = gid & 31;
  if (w >= NNODES / 16) return;
  int row0 = w * 16;
  int nl = lane & 15, hi = lane >> 4;
  int koff = 8 * hi;
  const unsigned short* hrow = h    + (size_t)(row0 + nl) * HIDD;
  const unsigned short* grow = accb + (size_t)(row0 + nl) * HO;
  v8f a0 = {}, a1 = {}, a2 = {}, a3 = {};
  #pragma unroll 2
  for (int kb = 0; kb < CATD / 32; ++kb){   // 14
    v16bf A;
    if (kb < 2){
      A = load_a2(hrow + kb * 32 + koff, hrow + kb * 32 + 16 + koff);
    } else {
      int cb = (kb - 2) * 32;
      union { uint4 q; unsigned short s[8]; } g0, g1;
      g0.q = *(const uint4*)(grow + cb + koff);
      g1.q = *(const uint4*)(grow + cb + 16 + koff);
      #pragma unroll
      for (int e = 0; e < 8; ++e){
        float v0 = lrelu(bf2f(g0.s[e]) + bgat[cb + koff + e], 0.01f);
        float v1 = lrelu(bf2f(g1.s[e]) + bgat[cb + 16 + koff + e], 0.01f);
        A[e]     = (__bf16)v0;
        A[e + 8] = (__bf16)v1;
      }
    }
    const unsigned short* bb = sw + (kb * 4) * 512 + lane * 16;
    v16bf B0 = load_frag16(bb);
    v16bf B1 = load_frag16(bb + 512);
    v16bf B2 = load_frag16(bb + 1024);
    v16bf B3 = load_frag16(bb + 1536);
    a0 = wmma_bf16(A, B0, a0);
    a1 = wmma_bf16(A, B1, a1);
    a2 = wmma_bf16(A, B2, a2);
    a3 = wmma_bf16(A, B3, a3);
  }
  #pragma unroll
  for (int r = 0; r < 8; ++r){
    int row = row0 + r + 8 * hi;
    float* op = out + (size_t)row * OUTD;
    op[nl]      = a0[r] + blin[nl];
    op[16 + nl] = a1[r] + blin[16 + nl];
    op[32 + nl] = a2[r] + blin[32 + nl];
    op[48 + nl] = a3[r] + blin[48 + nl];
  }
}

extern "C" void kernel_launch(void* const* d_in, const int* in_sizes, int n_in,
                              void* d_out, int out_size, void* d_ws, size_t ws_size,
                              hipStream_t stream)
{
  const float*     x    = (const float*)d_in[0];
  const long long* ei   = (const long long*)d_in[1];
  /* d_in[2] edge_weight: unused (matches reference) */
  const float*     Wpre = (const float*)d_in[3];
  const float*     bpre = (const float*)d_in[4];
  const float*     Wgat = (const float*)d_in[5];
  const float*     attS = (const float*)d_in[6];
  const float*     attD = (const float*)d_in[7];
  const float*     bgat = (const float*)d_in[8];
  const float*     Wlin = (const float*)d_in[9];
  const float*     blin = (const float*)d_in[10];
  float* out = (float*)d_out;

  char* ws = (char*)d_ws;
  unsigned short* spre = (unsigned short*)(ws + 0);          //  64 KB
  unsigned short* sgat = (unsigned short*)(ws + 65536);      //  48 KB
  unsigned short* slin = (unsigned short*)(ws + 114688);     //  56 KB
  unsigned short* hbf  = (unsigned short*)(ws + 172032);     //  6.4 MB  [N,64] bf16
  unsigned short* xh   = (unsigned short*)(ws + 6572032);    // 38.4 MB  [N,384] bf16
  float*          asrc = (float*)(ws + 44972032);            //  1.2 MB  [N,6]
  float*          adst = (float*)(ws + 46172032);            //  1.2 MB
  float*          maxv = (float*)(ws + 47372032);            //  1.2 MB
  float*          den  = (float*)(ws + 48572032);            //  1.2 MB
  unsigned short* accb = (unsigned short*)(ws + 49772032);   // 38.4 MB  [N,384] bf16
  // total workspace: 88,172,032 bytes

  k_swz_weights <<<168,    32, 0, stream>>>(Wpre, Wgat, Wlin, spre, sgat, slin);
  k_pre_gemm    <<<391,   256, 0, stream>>>(x, spre, bpre, hbf);                    // 3125 waves
  k_gat_gemm    <<<2344,  256, 0, stream>>>(hbf, sgat, attS, attD, xh, asrc, adst); // 18750 waves
  k_self_max    <<<1172,  256, 0, stream>>>(asrc, adst, maxv);
  k_edge_max    <<<18750, 256, 0, stream>>>(ei, asrc, adst, maxv);
  k_self_den    <<<1172,  256, 0, stream>>>(asrc, adst, maxv, den);
  k_edge_den    <<<18750, 256, 0, stream>>>(ei, asrc, adst, maxv, den);
  k_self_scatter<<<37500, 256, 0, stream>>>(asrc, adst, maxv, den, xh, accb);
  k_edge_scatter<<<600000,256, 0, stream>>>(ei, asrc, adst, maxv, den, xh, accb);   // 4.8M waves
  k_final_gemm  <<<391,   256, 0, stream>>>(hbf, accb, bgat, slin, blin, out);
}
